// TorchMACE_16819091931702
// MI455X (gfx1250) — matrix operand order repaired and checked by reading, not compiled, not run on hardware
//
#include <hip/hip_runtime.h>
#include <math.h>

#define EDGES 8192
#define NODES 1024
#define K1TOT 8320   // 65 * 128 (64 j-rows + 1 bias row) * U=128
#define K2TOT 4160   // 65 * 64

typedef _Float16 f16_t;
typedef __attribute__((ext_vector_type(16))) _Float16 v16h;
typedef __attribute__((ext_vector_type(8)))  float    v8f;

#define SH0C  0.28209479177387814f
#define SH1C  0.4886025119029199f
#define INV3C 0.57735026918962576f
#define PW0   0.08838834764831845f   // 1/sqrt(128)
#define PWL   0.07216878364870323f   // 1/sqrt(192)
#define ISQHS 0.08838834764831845f
#define ISQHV 0.125f

__device__ __forceinline__ float siluf(float x) { return x / (1.0f + expf(-x)); }

// ---------------------------------------------------------------- utility ----
__global__ void mace_zero(float* p, int n) {
    int i = blockIdx.x * blockDim.x + threadIdx.x;
    if (i < n) p[i] = 0.0f;
}

// ------------------------------------------------------------ edge geometry --
__global__ void mace_edge_geom(const float* __restrict__ pos, const int* __restrict__ eidx,
                               const float* __restrict__ shifts,
                               float* __restrict__ sh1, float* __restrict__ gate,
                               float* __restrict__ eattr) {
    int e = blockIdx.x * blockDim.x + threadIdx.x;
    if (e >= EDGES) return;
    int r = eidx[e], c = eidx[EDGES + e];
    float v0 = pos[c * 3 + 0] - pos[r * 3 + 0] + shifts[e * 3 + 0];
    float v1 = pos[c * 3 + 1] - pos[r * 3 + 1] + shifts[e * 3 + 1];
    float v2 = pos[c * 3 + 2] - pos[r * 3 + 2] + shifts[e * 3 + 2];
    float d = sqrtf(v0 * v0 + v1 * v1 + v2 * v2);
    float inv = 1.0f / fmaxf(d, 1e-8f);
    sh1[e * 3 + 0] = SH1C * v0 * inv;
    sh1[e * 3 + 1] = SH1C * v1 * inv;
    sh1[e * 3 + 2] = SH1C * v2 * inv;
    float s = d * 0.2f;
    float g = 0.0f;
    if (s < 1.0f) { float den = 1.0f - s * s; g = expf(1.0f - 1.0f / den); }
    gate[e] = g;
    const float gamma = 38.44f;  // (31/5)^2
    for (int b = 0; b < 32; ++b) {
        float diff = d - (5.0f / 31.0f) * (float)b;
        eattr[e * 32 + b] = expf(-gamma * diff * diff) * g;
    }
}

// ---------------------------------------------------------------- embedding --
__global__ void mace_embed(const int* __restrict__ z, const float* __restrict__ embed,
                           float* __restrict__ xs) {
    int i = blockIdx.x * blockDim.x + threadIdx.x;
    if (i >= NODES * 128) return;
    int n = i >> 7, u = i & 127;
    xs[i] = embed[z[n] * 128 + u];
}

// --------------------------------------------------------------- radial MLP --
__global__ void mace_radial(const float* __restrict__ eattr, const float* __restrict__ rw1,
                            const float* __restrict__ rb1, f16_t* __restrict__ Hf) {
    int i = blockIdx.x * blockDim.x + threadIdx.x;
    if (i >= EDGES * 64) return;
    int e = i >> 6, j = i & 63;
    float acc = rb1[j];
    for (int b = 0; b < 32; ++b) acc += eattr[e * 32 + b] * rw1[b * 64 + j];
    Hf[i] = (f16_t)siluf(acc);
}

// --------------------------------------- permute rw2 -> N-major f16 B^T mats --
// Bt1 rows: n<128 -> path a (0e*0e->0e), n>=128 -> path b (0e*1o->1o). K index
// kk = j*128+u for j<64; kk>=8192 is the rb2 bias region (virtual j==64, h=1).
__global__ void mace_permute1(const float* __restrict__ rw2, const float* __restrict__ rb2,
                              int numel, f16_t* __restrict__ Bt1) {
    int i = blockIdx.x * blockDim.x + threadIdx.x;
    if (i >= 192 * K1TOT) return;
    int n = i / K1TOT, kk = i - n * K1TOT;
    int j = 0, u;
    bool bias = (kk >= 8192);
    if (!bias) { j = kk >> 7; u = kk & 127; } else { u = kk - 8192; }
    int off = (n < 128) ? (u * 128 + n) : (16384 + u * 64 + (n - 128));
    float val = bias ? rb2[off] : rw2[(size_t)j * numel + off];
    Bt1[i] = (f16_t)val;
}

// Bt2 rows: n<64 -> path c (1o*0e->1o), n>=64 -> path d (1o*1o->0e).
__global__ void mace_permute2(const float* __restrict__ rw2, const float* __restrict__ rb2,
                              int numel, f16_t* __restrict__ Bt2) {
    int i = blockIdx.x * blockDim.x + threadIdx.x;
    if (i >= 192 * K2TOT) return;
    int n = i / K2TOT, kk = i - n * K2TOT;
    int j = 0, u;
    bool bias = (kk >= 4096);
    if (!bias) { j = kk >> 6; u = kk & 63; } else { u = kk - 4096; }
    int off = (n < 64) ? (24576 + u * 64 + n) : (28672 + u * 128 + (n - 64));
    float val = bias ? rb2[off] : rw2[(size_t)j * numel + off];
    Bt2[i] = (f16_t)val;
}

// --------------------------- gather sender features per edge, convert to f16 --
__global__ void mace_edge_prep(const int* __restrict__ eidx, const float* __restrict__ xs,
                               const float* __restrict__ xv, const float* __restrict__ sh1,
                               f16_t* __restrict__ Xab, f16_t* __restrict__ Xcd, int hasV) {
    int i = blockIdx.x * blockDim.x + threadIdx.x;
    if (i >= EDGES * 128) return;
    int e = i >> 7, u = i & 127;
    int c = eidx[EDGES + e];
    Xab[i] = (f16_t)xs[c * 128 + u];
    if (hasV && u < 64) {
        float a0 = xv[(c * 64 + u) * 3 + 0];
        float a1 = xv[(c * 64 + u) * 3 + 1];
        float a2 = xv[(c * 64 + u) * 3 + 2];
        Xcd[(size_t)(0 * EDGES + e) * 64 + u] = (f16_t)a0;
        Xcd[(size_t)(1 * EDGES + e) * 64 + u] = (f16_t)a1;
        Xcd[(size_t)(2 * EDGES + e) * 64 + u] = (f16_t)a2;
        float vd = a0 * sh1[e * 3 + 0] + a1 * sh1[e * 3 + 1] + a2 * sh1[e * 3 + 2];
        Xcd[(size_t)(3 * EDGES + e) * 64 + u] = (f16_t)vd;
    }
}

// -------------------------------------------------- fused outer-product GEMM --
// C[r,n] = sum_{j<=64, u<U} (h[r,j] * X[r,u]) * Bt[n, j*U+u], where h[r,64]:=1
// implements the rb2 bias rows.  Block = 128 threads = 4 waves; block tile is
// 64 rows x 64 cols: wave w owns rows w*16..w*16+15 and all 4 column subtiles
// (4 v8f accumulators, 4 v_wmma_f32_16x16x32_f16 per 32-wide K chunk, A
// fragment amortized 4x, per-j h value amortized over U/32 chunks).
template <int USHIFT>
__global__ __launch_bounds__(128) void mace_gemm(
    const f16_t* __restrict__ X, const f16_t* __restrict__ H, int eMask,
    const f16_t* __restrict__ Bt, float* __restrict__ C, int ldc) {
    constexpr int U = 1 << USHIFT;
    constexpr int K = 65 * U;
    constexpr int NCH = U / 32;  // K chunks per j
    __shared__ __align__(16) f16_t Xs[64][136];  // 272B stride: conflict-free, 16B aligned
    __shared__ __align__(16) f16_t Hs[64][72];   // col 64 = ones (bias row)
    const int tid = threadIdx.x;
    const int rowBase = blockIdx.x * 64;

    for (int idx = tid; idx < 64 * (U / 8); idx += 128) {
        int r = idx / (U / 8), c8 = idx % (U / 8);
        *(uint4*)&Xs[r][c8 * 8] = *(const uint4*)(X + (size_t)(rowBase + r) * U + c8 * 8);
    }
    for (int idx = tid; idx < 64 * 8; idx += 128) {
        int r = idx >> 3, c8 = idx & 7;
        *(uint4*)&Hs[r][c8 * 8] =
            *(const uint4*)(H + (size_t)((rowBase + r) & eMask) * 64 + c8 * 8);
    }
    for (int r = tid; r < 64; r += 128) Hs[r][64] = (f16_t)1.0f;
    __syncthreads();

    const int lane = tid & 31;
    const int wave = tid >> 5;
    const int half = lane >> 4;
    const int l16 = lane & 15;
    const int m = wave * 16 + l16;
    const int colBase = blockIdx.y * 64;

    v8f acc[4];
#pragma unroll
    for (int s = 0; s < 4; ++s) acc[s] = (v8f){0.f, 0.f, 0.f, 0.f, 0.f, 0.f, 0.f, 0.f};
    const f16_t* bp0 = Bt + (size_t)(colBase + 0 * 16 + l16) * K + half * 16;
    const f16_t* bp1 = Bt + (size_t)(colBase + 1 * 16 + l16) * K + half * 16;
    const f16_t* bp2 = Bt + (size_t)(colBase + 2 * 16 + l16) * K + half * 16;
    const f16_t* bp3 = Bt + (size_t)(colBase + 3 * 16 + l16) * K + half * 16;
    const f16_t* xrow = &Xs[m][half * 16];

    for (int j = 0; j <= 64; ++j) {
        f16_t hh = Hs[m][j];
#pragma unroll
        for (int c = 0; c < NCH; ++c) {
            const f16_t* xr = xrow + c * 32;
            v16h a;
#pragma unroll
            for (int i = 0; i < 8; ++i) {  // A layout: lanes0-15 K{0..15}; lanes16-31 K{16..31}
                a[i]     = xr[i]      * hh;
                a[i + 8] = xr[i + 8] * hh;
            }
            const int kc = j * U + c * 32;
            __builtin_prefetch(bp0 + kc + 1024, 0, 1);  // speculative; stays inside d_ws
            v16h b0 = *(const v16h*)(bp0 + kc);  // B layout: lane n = lane&15, K = half*16+i
            v16h b1 = *(const v16h*)(bp1 + kc);
            v16h b2 = *(const v16h*)(bp2 + kc);
            v16h b3 = *(const v16h*)(bp3 + kc);
            acc[0] = __builtin_amdgcn_wmma_f32_16x16x32_f16(false, a, false, b0, (short)0, acc[0], false, false);
            acc[1] = __builtin_amdgcn_wmma_f32_16x16x32_f16(false, a, false, b1, (short)0, acc[1], false, false);
            acc[2] = __builtin_amdgcn_wmma_f32_16x16x32_f16(false, a, false, b2, (short)0, acc[2], false, false);
            acc[3] = __builtin_amdgcn_wmma_f32_16x16x32_f16(false, a, false, b3, (short)0, acc[3], false, false);
        }
    }
    const int mOut = rowBase + wave * 16 + half * 8;
#pragma unroll
    for (int s = 0; s < 4; ++s) {
        const int n = colBase + s * 16 + l16;
#pragma unroll
        for (int r = 0; r < 8; ++r) C[(size_t)(mOut + r) * ldc + n] = acc[s][r];
    }
}

// ------------------------------------- message epilogue + segment scatter-add --
__global__ void mace_scatter(const float* __restrict__ S1, const float* __restrict__ Sc,
                             const float* __restrict__ Sd, const float* __restrict__ sh1,
                             const float* __restrict__ gate, const int* __restrict__ eidx,
                             float pw, int hasV, float* __restrict__ agg0,
                             float* __restrict__ agg1) {
    int i = blockIdx.x * blockDim.x + threadIdx.x;
    if (i >= EDGES * 128) return;
    int e = i >> 7, w = i & 127;
    float gpw = gate[e] * pw;
    int dst = eidx[e];  // row = receiver
    float m0 = SH0C * S1[(size_t)e * 192 + w];
    if (hasV) m0 += INV3C * Sd[(size_t)e * 128 + w];
    atomicAdd(&agg0[dst * 128 + w], m0 * gpw);
    if (w < 64) {
        float sb = S1[(size_t)e * 192 + 128 + w];
        for (int k = 0; k < 3; ++k) {
            float m1 = INV3C * sb * sh1[e * 3 + k];
            if (hasV) m1 += SH0C * Sc[((size_t)k * EDGES + e) * 64 + w];
            atomicAdd(&agg1[(dst * 64 + w) * 3 + k], m1 * gpw);
        }
    }
}

// ----------------------------------------------- node update: self-interaction --
__global__ void mace_update1(const float* __restrict__ agg0, const float* __restrict__ agg1,
                             const float* __restrict__ xs, const float* __restrict__ xv,
                             const float* __restrict__ si_s, const float* __restrict__ si_v,
                             int hasV, float* __restrict__ xsB, float* __restrict__ xvB) {
    int i = blockIdx.x * blockDim.x + threadIdx.x;
    if (i >= NODES * 128) return;
    int n = i >> 7, w = i & 127;
    float s = 0.0f;
    for (int u = 0; u < 128; ++u) s += xs[n * 128 + u] * si_s[u * 128 + w];
    xsB[i] = agg0[i] + s * ISQHS;
    if (w < 64) {
        for (int k = 0; k < 3; ++k) {
            float a = agg1[(n * 64 + w) * 3 + k];
            if (hasV) {
                float sv = 0.0f;
                for (int u = 0; u < 64; ++u) sv += xv[(n * 64 + u) * 3 + k] * si_v[u * 64 + w];
                a += sv * ISQHV;
            }
            xvB[(n * 64 + w) * 3 + k] = a;
        }
    }
}

__global__ void mace_mlp1(const float* __restrict__ xsB, const float* __restrict__ mw1,
                          const float* __restrict__ mb1, float* __restrict__ t1) {
    int i = blockIdx.x * blockDim.x + threadIdx.x;
    if (i >= NODES * 128) return;
    int n = i >> 7, j = i & 127;
    float acc = mb1[j];
    for (int u = 0; u < 128; ++u) acc += xsB[n * 128 + u] * mw1[u * 128 + j];
    t1[i] = siluf(acc);
}

__global__ void mace_mlp2(const float* __restrict__ xsB, const float* __restrict__ t1,
                          const float* __restrict__ mw2, const float* __restrict__ mb2,
                          float* __restrict__ xsOut) {
    int i = blockIdx.x * blockDim.x + threadIdx.x;
    if (i >= NODES * 128) return;
    int n = i >> 7, w = i & 127;
    float acc = mb2[w];
    for (int j = 0; j < 128; ++j) acc += t1[n * 128 + j] * mw2[j * 128 + w];
    xsOut[i] = xsB[i] + acc;
}

// --------------------------------------------------------------------- readout --
__global__ void mace_ln(const float* __restrict__ xs, float* __restrict__ mu,
                        float* __restrict__ rstd) {
    int n = blockIdx.x * blockDim.x + threadIdx.x;
    if (n >= NODES) return;
    float m = 0.0f;
    for (int w = 0; w < 128; ++w) m += xs[n * 128 + w];
    m *= (1.0f / 128.0f);
    float v = 0.0f;
    for (int w = 0; w < 128; ++w) { float d = xs[n * 128 + w] - m; v += d * d; }
    v *= (1.0f / 128.0f);
    mu[n] = m;
    rstd[n] = rsqrtf(v + 1e-5f);
}

__global__ void mace_ro1(const float* __restrict__ xs, const float* __restrict__ mu,
                         const float* __restrict__ rstd, const float* __restrict__ g,
                         const float* __restrict__ b, const float* __restrict__ w1,
                         const float* __restrict__ b1, float* __restrict__ t2) {
    int i = blockIdx.x * blockDim.x + threadIdx.x;
    if (i >= NODES * 64) return;
    int n = i >> 6, j = i & 63;
    float acc = b1[j];
    float m = mu[n], rs = rstd[n];
    for (int w = 0; w < 128; ++w) {
        float hn = (xs[n * 128 + w] - m) * rs * g[w] + b[w];
        acc += hn * w1[w * 64 + j];
    }
    t2[i] = siluf(acc);
}

__global__ void mace_ro2(const float* __restrict__ t2, const float* __restrict__ w2,
                         const float* __restrict__ b2, const int* __restrict__ batch,
                         float* __restrict__ out) {
    int n = blockIdx.x * blockDim.x + threadIdx.x;
    if (n >= NODES) return;
    float acc = b2[0];
    for (int j = 0; j < 64; ++j) acc += t2[n * 64 + j] * w2[j];
    atomicAdd(&out[batch[n]], acc);
}

// ================================================================== launcher ==
static inline char* mace_carve(char*& p, size_t bytes) {
    char* r = p;
    p += (bytes + 255) & ~(size_t)255;
    return r;
}

extern "C" void kernel_launch(void* const* d_in, const int* in_sizes, int n_in,
                              void* d_out, int out_size, void* d_ws, size_t ws_size,
                              hipStream_t stream) {
    int p = 0;
    const int*   z      = (const int*)d_in[p++];
    const float* pos    = (const float*)d_in[p++];
    const int*   eidx   = (const int*)d_in[p++];
    const float* shifts = (const float*)d_in[p++];
    const int*   batch  = (const int*)d_in[p++];
    p++;  // num_graphs scalar (unused on host)
    const float* embed  = (const float*)d_in[p++];
    const float *rw1[4], *rb1[4], *rw2[4], *rb2[4], *si_s[4], *mw1[4], *mb1[4], *mw2[4],
                *mb2[4], *si_v[4];
    for (int l = 0; l < 4; ++l) {
        rw1[l]  = (const float*)d_in[p++]; rb1[l] = (const float*)d_in[p++];
        rw2[l]  = (const float*)d_in[p++]; rb2[l] = (const float*)d_in[p++];
        si_s[l] = (const float*)d_in[p++];
        mw1[l]  = (const float*)d_in[p++]; mb1[l] = (const float*)d_in[p++];
        mw2[l]  = (const float*)d_in[p++]; mb2[l] = (const float*)d_in[p++];
        si_v[l] = (l > 0) ? (const float*)d_in[p++] : nullptr;
    }
    const float* ln_g  = (const float*)d_in[p++];
    const float* ln_b  = (const float*)d_in[p++];
    const float* ro_w1 = (const float*)d_in[p++];
    const float* ro_b1 = (const float*)d_in[p++];
    const float* ro_w2 = (const float*)d_in[p++];
    const float* ro_b2 = (const float*)d_in[p++];
    float* out = (float*)d_out;

    // ---- workspace carve ----
    char* w = (char*)d_ws;
    float* sh1   = (float*)mace_carve(w, (size_t)EDGES * 3 * 4);
    float* gate  = (float*)mace_carve(w, (size_t)EDGES * 4);
    float* eattr = (float*)mace_carve(w, (size_t)EDGES * 32 * 4);
    f16_t* Hf    = (f16_t*)mace_carve(w, (size_t)EDGES * 64 * 2);
    f16_t* Xab   = (f16_t*)mace_carve(w, (size_t)EDGES * 128 * 2);
    f16_t* Xcd   = (f16_t*)mace_carve(w, (size_t)4 * EDGES * 64 * 2);
    f16_t* Bt1   = (f16_t*)mace_carve(w, (size_t)192 * K1TOT * 2);
    f16_t* Bt2   = (f16_t*)mace_carve(w, (size_t)192 * K2TOT * 2);
    float* S1    = (float*)mace_carve(w, (size_t)EDGES * 192 * 4);
    float* Sc    = (float*)mace_carve(w, (size_t)3 * EDGES * 64 * 4);
    float* Sd    = (float*)mace_carve(w, (size_t)EDGES * 128 * 4);
    float* agg0  = (float*)mace_carve(w, (size_t)NODES * 128 * 4);
    float* agg1  = (float*)mace_carve(w, (size_t)NODES * 192 * 4);
    float* xsA   = (float*)mace_carve(w, (size_t)NODES * 128 * 4);
    float* xsB   = (float*)mace_carve(w, (size_t)NODES * 128 * 4);
    float* xvA   = (float*)mace_carve(w, (size_t)NODES * 192 * 4);
    float* xvB   = (float*)mace_carve(w, (size_t)NODES * 192 * 4);
    float* t1    = (float*)mace_carve(w, (size_t)NODES * 128 * 4);
    float* t2    = (float*)mace_carve(w, (size_t)NODES * 64 * 4);
    float* muB   = (float*)mace_carve(w, (size_t)NODES * 4);
    float* rsB   = (float*)mace_carve(w, (size_t)NODES * 4);
    if ((size_t)(w - (char*)d_ws) > ws_size) return;  // insufficient scratch

    auto blks = [](long n) { return dim3((unsigned)((n + 255) / 256)); };
    const int numel[4] = {24576, 36864, 36864, 36864};

    float* xv_cur = xvA;
    float* xv_nxt = xvB;

    mace_zero<<<blks(NODES * 192), 256, 0, stream>>>(xv_cur, NODES * 192);
    mace_edge_geom<<<blks(EDGES), 256, 0, stream>>>(pos, eidx, shifts, sh1, gate, eattr);
    mace_embed<<<blks(NODES * 128), 256, 0, stream>>>(z, embed, xsA);

    for (int l = 0; l < 4; ++l) {
        int hasV = (l > 0) ? 1 : 0;
        mace_radial<<<blks(EDGES * 64), 256, 0, stream>>>(eattr, rw1[l], rb1[l], Hf);
        mace_permute1<<<blks((long)192 * K1TOT), 256, 0, stream>>>(rw2[l], rb2[l], numel[l], Bt1);
        if (hasV)
            mace_permute2<<<blks((long)192 * K2TOT), 256, 0, stream>>>(rw2[l], rb2[l], numel[l], Bt2);
        mace_edge_prep<<<blks(EDGES * 128), 256, 0, stream>>>(eidx, xsA, xv_cur, sh1, Xab, Xcd, hasV);

        // GEMM1: [8192, 8320] x Bt1[192, 8320]^T -> S1[8192, 192]
        mace_gemm<7><<<dim3(EDGES / 64, 3), 128, 0, stream>>>(Xab, Hf, EDGES - 1, Bt1, S1, 192);
        if (hasV) {
            // GEMM2a: xv_{x,y,z} stacked along M: [24576, 4160] x Bt2c[64,4160]^T -> Sc
            mace_gemm<6><<<dim3(3 * EDGES / 64, 1), 128, 0, stream>>>(Xcd, Hf, EDGES - 1, Bt2,
                                                                      Sc, 64);
            // GEMM2b: vdots: [8192, 4160] x Bt2d[128,4160]^T -> Sd
            mace_gemm<6><<<dim3(EDGES / 64, 2), 128, 0, stream>>>(
                Xcd + (size_t)3 * EDGES * 64, Hf, EDGES - 1, Bt2 + (size_t)64 * K2TOT, Sd, 128);
        }

        mace_zero<<<blks(NODES * 128), 256, 0, stream>>>(agg0, NODES * 128);
        mace_zero<<<blks(NODES * 192), 256, 0, stream>>>(agg1, NODES * 192);
        mace_scatter<<<blks(EDGES * 128), 256, 0, stream>>>(S1, Sc, Sd, sh1, gate, eidx,
                                                            l ? PWL : PW0, hasV, agg0, agg1);
        mace_update1<<<blks(NODES * 128), 256, 0, stream>>>(agg0, agg1, xsA, xv_cur, si_s[l],
                                                            si_v[l], hasV, xsB, xv_nxt);
        mace_mlp1<<<blks(NODES * 128), 256, 0, stream>>>(xsB, mw1[l], mb1[l], t1);
        mace_mlp2<<<blks(NODES * 128), 256, 0, stream>>>(xsB, t1, mw2[l], mb2[l], xsA);
        float* tmp = xv_cur; xv_cur = xv_nxt; xv_nxt = tmp;
    }

    mace_ln<<<blks(NODES), 256, 0, stream>>>(xsA, muB, rsB);
    mace_ro1<<<blks(NODES * 64), 256, 0, stream>>>(xsA, muB, rsB, ln_g, ln_b, ro_w1, ro_b1, t2);
    mace_zero<<<1, 256, 0, stream>>>(out, out_size);
    mace_ro2<<<blks(NODES), 256, 0, stream>>>(t2, ro_w2, ro_b2, batch, out);
}
